// Embedding_layer_43533788512430
// MI455X (gfx1250) — matrix operand order repaired and checked by reading, not compiled, not run on hardware
//
#include <hip/hip_runtime.h>
#include <hip/hip_bf16.h>

#define EDIM    128
#define RDIM    64
#define NBR     6
#define TOKENS  51200      /* B*L = 256*200 */
#define NEG_BIG (-1e10f)
#define WPB     4          /* waves per block */

typedef __attribute__((ext_vector_type(16))) __bf16 v16bf;
typedef __attribute__((ext_vector_type(8)))  float  v8f;
typedef __attribute__((ext_vector_type(4)))  unsigned int u32x4;

__device__ __forceinline__ unsigned int f2bf1(float f) {
    unsigned int u = __builtin_bit_cast(unsigned int, f);
    return (u + 0x7FFFu + ((u >> 16) & 1u)) >> 16;   // round-to-nearest-even
}
__device__ __forceinline__ unsigned long long packbf4(float a, float b, float c, float d) {
    unsigned long long lo = (unsigned long long)(f2bf1(a) | (f2bf1(b) << 16));
    unsigned long long hi = (unsigned long long)(f2bf1(c) | (f2bf1(d) << 16));
    return lo | (hi << 32);
}
__device__ __forceinline__ unsigned int packbf2(float a, float b) {
    return f2bf1(a) | (f2bf1(b) << 16);
}

union CatMem {                         // per-wave 16x256 bf16 cat tile
    unsigned long long d[WPB][1024];   // [wave][row*64 + slot]
    u32x4              q[WPB][512];    // [wave][row*32 + q]  (row = 256 bf16 = 32 u32x4)
};
union ABv { v16bf v; u32x4 q[2]; };

// ---- prep: weff[k] = sum_r ua[r]*Wa_w[r,k] ; beff = ua.Wa_b  (one block) ----
__global__ void gat_prep_kernel(const float* __restrict__ Wa_w,
                                const float* __restrict__ Wa_b,
                                const float* __restrict__ ua,
                                float* __restrict__ ws)
{
    const int t = threadIdx.x;                      // 64 threads, 4 k's each
    float a0 = 0.f, a1 = 0.f, a2 = 0.f, a3 = 0.f;
    for (int r = 0; r < RDIM; ++r) {
        const float  u = ua[r];
        const float4 w = *(const float4*)(Wa_w + r * 256 + t * 4);
        a0 += u * w.x; a1 += u * w.y; a2 += u * w.z; a3 += u * w.w;
    }
    *(float4*)(ws + t * 4) = make_float4(a0, a1, a2, a3);
    if (t == 0) {
        float b = 0.f;
        for (int r = 0; r < RDIM; ++r) b += ua[r] * Wa_b[r];
        ws[256] = b;
    }
}

__launch_bounds__(32 * WPB)
__global__ void gat_attn_kernel(const int* __restrict__ xs,
                                const int* __restrict__ path_map,
                                const float* __restrict__ embed,
                                const float* __restrict__ ws,   // weff[256], beff
                                float* __restrict__ out)
{
    // B tile (256x16 bf16): col0 = weff, col1 = (k>=128 ? 1 : 0), cols 2..15 = 0
    __shared__ u32x4  BQ[512];                      // 16 cols x 32 u32x4
    __shared__ CatMem cat;
    __shared__ unsigned long long eibf[WPB][512];   // ei as bf16: [wave][row*32 + lane]
    __shared__ float  scors[WPB][16][NBR];
    __shared__ float  rsums[WPB][16][NBR];
    __shared__ int    nid_s[WPB][16][NBR];
    __shared__ int    xid_s[WPB][16];

    const int  tid  = threadIdx.x;
    const int  lane = tid & 31;
    const int  wv   = tid >> 5;
    const int  lh   = lane & 15;
    const bool hi   = lane >= 16;

    // ---- build B tile ----
    for (int i = tid; i < 512; i += 32 * WPB) {
        const int col = i >> 5;                     // 0..15
        const int qk  = i & 31;                     // u32x4 within column; k = qk*8
        u32x4 q = {0u, 0u, 0u, 0u};
        if (col == 0) {
            const float* p = ws + qk * 8;
            q.x = packbf2(p[0], p[1]);
            q.y = packbf2(p[2], p[3]);
            q.z = packbf2(p[4], p[5]);
            q.w = packbf2(p[6], p[7]);
        } else if (col == 1 && qk >= 16) {          // k in [128,256): bf16 1.0 pairs
            q.x = 0x3F803F80u; q.y = 0x3F803F80u; q.z = 0x3F803F80u; q.w = 0x3F803F80u;
        }
        BQ[i] = q;
    }

    const int tile    = blockIdx.x * WPB + wv;      // 0..3199
    const int tokBase = tile * 16;
    const float beff  = ws[256];

    if (lane < 16) xid_s[wv][lane] = xs[tokBase + lane];
    __syncthreads();

    for (int i = lane; i < 16 * NBR; i += 32) {
        int row = i / NBR, n = i % NBR;
        nid_s[wv][row][n] = path_map[xid_s[wv][row] * NBR + n];
    }
    __syncthreads();                                // BQ + nid_s visible

    // ---- cache ei rows once as bf16 (reused by all 6 neighbor passes) ----
    for (int row = 0; row < 16; ++row) {
        const float4 ei = *(const float4*)(embed + (size_t)xid_s[wv][row] * EDIM + lane * 4);
        eibf[wv][row * 32 + lane] = packbf4(ei.x, ei.y, ei.z, ei.w);
    }

    for (int n = 0; n < NBR; ++n) {
        // ---- stage cat tile: row = [where(ej==0,0,ei) | ej] as bf16 ----
        for (int row = 0; row < 16; ++row) {
            const float4 ej = *(const float4*)(embed + (size_t)nid_s[wv][row][n] * EDIM + lane * 4);
            unsigned int m0 = (ej.x == 0.f ? 0u : 0x0000FFFFu) | (ej.y == 0.f ? 0u : 0xFFFF0000u);
            unsigned int m1 = (ej.z == 0.f ? 0u : 0x0000FFFFu) | (ej.w == 0.f ? 0u : 0xFFFF0000u);
            unsigned long long msk = (unsigned long long)m0 | ((unsigned long long)m1 << 32);
            cat.d[wv][row * 64 + lane]      = eibf[wv][row * 32 + lane] & msk;  // k = lane*4
            cat.d[wv][row * 64 + 32 + lane] = packbf4(ej.x, ej.y, ej.z, ej.w);  // k = 128+lane*4
        }

        // prefetch next neighbor's rows into cache while WMMA runs
        if (n + 1 < NBR) {
            for (int row = 0; row < 16; ++row)
                __builtin_prefetch(embed + (size_t)nid_s[wv][row][n + 1] * EDIM + lane * 4, 0, 0);
        }

        // ---- score (col 0) and ej row-sum (col 1) in one WMMA chain ----
        v8f Ca = {}, Cb = {};
#pragma unroll 1
        for (int kc = 0; kc < 4; ++kc) {
            ABv a, b;
            // even K-step
            int aq = lh * 32 + (2 * kc) * 4 + (hi ? 1 : 0);
            int bq = lh * 32 + (2 * kc) * 4 + (hi ? 1 : 0);
            a.q[0] = cat.q[wv][aq];  a.q[1] = cat.q[wv][aq + 2];
            b.q[0] = BQ[bq];         b.q[1] = BQ[bq + 2];
            Ca = __builtin_amdgcn_wmma_f32_16x16x32_bf16(false, a.v, false, b.v, (short)0, Ca, false, false);
            // odd K-step (independent accumulator breaks the WMMA->WMMA chain)
            aq += 4; bq += 4;
            a.q[0] = cat.q[wv][aq];  a.q[1] = cat.q[wv][aq + 2];
            b.q[0] = BQ[bq];         b.q[1] = BQ[bq + 2];
            Cb = __builtin_amdgcn_wmma_f32_16x16x32_bf16(false, a.v, false, b.v, (short)0, Cb, false, false);
        }

        // lane N=0 holds the score, lane N=1 holds sum(ej) for row j / j+8
        if (lh < 2) {
#pragma unroll
            for (int j = 0; j < 8; ++j) {
                const float v   = Ca[j] + Cb[j];
                const int   row = hi ? (j + 8) : j;
                float* dst = (lh == 0) ? &scors[wv][row][n] : &rsums[wv][row][n];
                *dst = v;
            }
        }
    }
    __syncthreads();

    // ---- softmax over 6 neighbors + weighted neighbor sum (L2 re-gather) ----
    for (int row = 0; row < 16; ++row) {
        float s[NBR];
#pragma unroll
        for (int n = 0; n < NBR; ++n)
            s[n] = (rsums[wv][row][n] == 0.f) ? NEG_BIG : (scors[wv][row][n] + beff);
        float mx = fmaxf(fmaxf(fmaxf(s[0], s[1]), fmaxf(s[2], s[3])), fmaxf(s[4], s[5]));
        float e[NBR];
        float den = 0.f;
#pragma unroll
        for (int n = 0; n < NBR; ++n) { e[n] = __expf(s[n] - mx); den += e[n]; }
        const float inv = 1.f / den;
        float acc0 = 0.f, acc1 = 0.f, acc2 = 0.f, acc3 = 0.f;
#pragma unroll
        for (int n = 0; n < NBR; ++n) {
            const float4 ej = *(const float4*)(embed + (size_t)nid_s[wv][row][n] * EDIM + lane * 4);
            const float w = e[n] * inv;
            acc0 += w * ej.x; acc1 += w * ej.y; acc2 += w * ej.z; acc3 += w * ej.w;
        }
        *(float4*)(out + (size_t)(tokBase + row) * EDIM + lane * 4) =
            make_float4(acc0, acc1, acc2, acc3);
    }
}

extern "C" void kernel_launch(void* const* d_in, const int* in_sizes, int n_in,
                              void* d_out, int out_size, void* d_ws, size_t ws_size,
                              hipStream_t stream) {
    (void)in_sizes; (void)n_in; (void)out_size; (void)ws_size;
    const int*   xs       = (const int*)  d_in[0];
    const int*   path_map = (const int*)  d_in[1];
    const float* embed    = (const float*)d_in[2];
    const float* Wa_w     = (const float*)d_in[3];
    const float* Wa_b     = (const float*)d_in[4];
    const float* ua       = (const float*)d_in[5];
    float*       ws       = (float*)      d_ws;     // weff[256] + beff
    float*       out      = (float*)      d_out;

    gat_prep_kernel<<<dim3(1), dim3(64), 0, stream>>>(Wa_w, Wa_b, ua, ws);
    // 16 tokens per wave, WPB waves per block
    gat_attn_kernel<<<dim3(TOKENS / (16 * WPB)), dim3(32 * WPB), 0, stream>>>(
        xs, path_map, embed, ws, out);
}
